// Decoder_73143293051533
// MI455X (gfx1250) — matrix operand order repaired and checked by reading, not compiled, not run on hardware
//
#include <hip/hip_runtime.h>

#define V_    8
#define DU_   512
#define B_    256
#define TIN_  512
#define TDEC_ 127

typedef __attribute__((ext_vector_type(16))) __bf16 v16bf;
typedef __attribute__((ext_vector_type(8)))  __bf16 v8bf;
typedef __attribute__((ext_vector_type(8)))  float  v8f;

static __device__ inline float sigf(float x){ return 1.f/(1.f+__expf(-x)); }

// ---------------- one-time conversion kernels ----------------
__global__ __launch_bounds__(256) void k_conv(const float* __restrict__ s,
                                              __bf16* __restrict__ d, long n){
  long i = (long)blockIdx.x*blockDim.x + threadIdx.x;
  long stride = (long)gridDim.x*blockDim.x;
  for (; i < n; i += stride) d[i] = (__bf16)s[i];
}

// Wt[n*K + k] = W[(k+rowOff)*N + n]  (fp32 [K,N] row-major -> bf16 [N,K])
__global__ __launch_bounds__(256) void k_conv_t(const float* __restrict__ W,
                                                __bf16* __restrict__ Wt,
                                                int K, int N, int rowOff){
  int idx = blockIdx.x*256 + threadIdx.x;
  if (idx >= K*N) return;
  int k = idx / N, n = idx - k*N;
  Wt[(long)n*K + k] = (__bf16)W[(long)(k + rowOff)*N + n];
}

// ------- fused LSTM layer: z = A1@W1 + A2@W2 + b (+W0 row gather); gates in epilogue
// Wave N-fragments = same 16 gate-d columns of all FOUR gates (d, d+512, d+1024, d+1536),
// so each accumulator element holds (i,f,g,o) for one (row,d) -> gates fused, no z buffer.
// Block = 8 waves (2 M x 4 D), block tile 64 rows x 64 gate-d. Grid (512/64, 256/64).
__global__ __launch_bounds__(256) void k_lstm(
    const __bf16* __restrict__ A1, const __bf16* __restrict__ W1t,
    const __bf16* __restrict__ A2, const __bf16* __restrict__ W2t,
    const float* __restrict__ bias,
    const float* __restrict__ tokW, const int* __restrict__ tokIdx, int tokStride,
    float* __restrict__ c, __bf16* __restrict__ hbf)
{
  const int K = 512;
  int lane = threadIdx.x & 31, wave = threadIdx.x >> 5;
  int wm = wave >> 2, wn = wave & 3;
  int half = lane >> 4, l16 = lane & 15;
  int m0 = blockIdx.y*64 + wm*32;
  int d0 = blockIdx.x*64 + wn*16;          // gate-dim base for this wave

  v8f acc[2][4] = {};
  for (int pass = 0; pass < 2; ++pass){
    const __bf16* A  = pass ? A2  : A1;
    const __bf16* Wt = pass ? W2t : W1t;
    for (int k0 = 0; k0 < K; k0 += 32){
      v16bf a[2], b[4];
      #pragma unroll
      for (int fm = 0; fm < 2; ++fm){
        const __bf16* ap = A + (long)(m0 + fm*16 + l16)*K + k0 + half*16;
        v8bf lo = *(const v8bf*)ap;
        v8bf hi = *(const v8bf*)(ap + 8);
        a[fm] = __builtin_shufflevector(lo, hi, 0,1,2,3,4,5,6,7,8,9,10,11,12,13,14,15);
      }
      #pragma unroll
      for (int g = 0; g < 4; ++g)
        b[g] = *(const v16bf*)(Wt + (long)(g*512 + d0 + l16)*K + k0 + half*16);
      #pragma unroll
      for (int fm = 0; fm < 2; ++fm)
        #pragma unroll
        for (int g = 0; g < 4; ++g)
          acc[fm][g] = __builtin_amdgcn_wmma_f32_16x16x32_bf16(
              false, a[fm], false, b[g], (short)0, acc[fm][g], false, false);
    }
  }

  int d = d0 + l16;
  float bi = bias[d], bff = bias[512+d], bgg = bias[1024+d], bo = bias[1536+d];
  #pragma unroll
  for (int fm = 0; fm < 2; ++fm){
    #pragma unroll
    for (int i = 0; i < 8; ++i){
      int r = m0 + fm*16 + half*8 + i;
      float zi = acc[fm][0][i] + bi;
      float zf = acc[fm][1][i] + bff;
      float zg = acc[fm][2][i] + bgg;
      float zo = acc[fm][3][i] + bo;
      if (tokW){
        const float* trow = tokW + (long)tokIdx[(long)r*tokStride]*2048;
        zi += trow[d]; zf += trow[512+d]; zg += trow[1024+d]; zo += trow[1536+d];
      }
      long ci = (long)r*512 + d;
      float cn = sigf(zf)*c[ci] + sigf(zi)*tanhf(zg);
      c[ci] = cn;
      hbf[ci] = (__bf16)(sigf(zo)*tanhf(cn));
    }
  }
}

// ---------------- plain bf16 WMMA GEMM (bf16 out): C = A1@W1 (+A2@W2) ----------------
__global__ __launch_bounds__(256) void k_gemm(
    const __bf16* __restrict__ A1, const __bf16* __restrict__ W1t, int ldw1,
    const __bf16* __restrict__ A2, const __bf16* __restrict__ W2t, int ldw2,
    __bf16* __restrict__ Cbf, int N)
{
  const int K = 512;
  int lane = threadIdx.x & 31, wave = threadIdx.x >> 5;
  int wm = wave >> 2, wn = wave & 3;
  int half = lane >> 4, l16 = lane & 15;
  int m0 = blockIdx.y*64  + wm*32;
  int n0 = blockIdx.x*256 + wn*64;

  v8f acc[2][4] = {};
  for (int pass = 0; pass < 2; ++pass){
    const __bf16* A  = pass ? A2  : A1;
    const __bf16* Wt = pass ? W2t : W1t;
    int ldw          = pass ? ldw2 : ldw1;
    if (!A) continue;
    for (int k0 = 0; k0 < K; k0 += 32){
      v16bf a[2], b[4];
      #pragma unroll
      for (int fm = 0; fm < 2; ++fm){
        const __bf16* ap = A + (long)(m0 + fm*16 + l16)*K + k0 + half*16;
        v8bf lo = *(const v8bf*)ap;
        v8bf hi = *(const v8bf*)(ap + 8);
        a[fm] = __builtin_shufflevector(lo, hi, 0,1,2,3,4,5,6,7,8,9,10,11,12,13,14,15);
      }
      #pragma unroll
      for (int fn = 0; fn < 4; ++fn)
        b[fn] = *(const v16bf*)(Wt + (long)(n0 + fn*16 + l16)*ldw + k0 + half*16);
      #pragma unroll
      for (int fm = 0; fm < 2; ++fm)
        #pragma unroll
        for (int fn = 0; fn < 4; ++fn)
          acc[fm][fn] = __builtin_amdgcn_wmma_f32_16x16x32_bf16(
              false, a[fm], false, b[fn], (short)0, acc[fm][fn], false, false);
    }
  }
  #pragma unroll
  for (int fm = 0; fm < 2; ++fm)
    #pragma unroll
    for (int fn = 0; fn < 4; ++fn){
      int col = n0 + fn*16 + l16;
      #pragma unroll
      for (int i = 0; i < 8; ++i){
        int row = m0 + fm*16 + half*8 + i;
        Cbf[(long)row*N + col] = (__bf16)acc[fm][fn][i];
      }
    }
}

// ---------------- fused attention: scores -> softmax -> context ----------------
__global__ __launch_bounds__(512) void k_attn(const __bf16* __restrict__ qbf,
                                              const __bf16* __restrict__ mem,
                                              __bf16* __restrict__ ctxbf){
  __shared__ float qs[512];
  __shared__ float pw[512];
  __shared__ float tmp[512];
  int b = blockIdx.x, tid = threadIdx.x;
  qs[tid] = (float)qbf[(long)b*512 + tid];
  __syncthreads();

  // phase 1: score for t = tid (dot over DU=512, bf16 pairs)
  const unsigned* r = (const unsigned*)(mem + ((long)b*512 + tid)*512);
  float acc = 0.f;
  #pragma unroll 8
  for (int i = 0; i < 256; ++i){
    unsigned u = r[i];
    acc += __uint_as_float(u << 16)         * qs[2*i]
         + __uint_as_float(u & 0xFFFF0000u) * qs[2*i+1];
  }
  tmp[tid] = acc; __syncthreads();
  for (int s = 256; s > 0; s >>= 1){
    if (tid < s) tmp[tid] = fmaxf(tmp[tid], tmp[tid+s]);
    __syncthreads();
  }
  float mx = tmp[0]; __syncthreads();
  float p = __expf(acc - mx);
  pw[tid] = p; tmp[tid] = p; __syncthreads();
  for (int s = 256; s > 0; s >>= 1){
    if (tid < s) tmp[tid] += tmp[tid+s];
    __syncthreads();
  }
  float inv = 1.f/tmp[0];

  // phase 2: ctx[e=tid] = sum_t p[t]*mem[b,t,e]  (coalesced across threads)
  float a = 0.f;
  const __bf16* col = mem + (long)b*512*512 + tid;
  #pragma unroll 4
  for (int t2 = 0; t2 < 512; ++t2){
    if ((t2 & 15) == 0) __builtin_prefetch(col + (long)(t2+16)*512, 0, 0);
    a += pw[t2]*(float)col[(long)t2*512];
  }
  ctxbf[(long)b*512 + tid] = (__bf16)(a*inv);
}

// ---------------- tiny output layer: logits = attn@Wf + bf ----------------
__global__ __launch_bounds__(256) void k_logits(const __bf16* __restrict__ attnbf,
                                                const float* __restrict__ Wf,
                                                const float* __restrict__ bfv,
                                                float* __restrict__ out, int t){
  int idx = blockIdx.x*256 + threadIdx.x;    // 0..2047
  int b = idx >> 3, v = idx & 7;
  const __bf16* ar = attnbf + (long)b*512;
  float a = bfv[v];
  for (int k = 0; k < 512; ++k) a += (float)ar[k]*Wf[(long)k*8 + v];
  out[((long)b*TDEC_ + t)*V_ + v] = a;
}

extern "C" void kernel_launch(void* const* d_in, const int* in_sizes, int n_in,
                              void* d_out, int out_size, void* d_ws, size_t ws_size,
                              hipStream_t stream){
  (void)in_sizes; (void)n_in; (void)out_size; (void)ws_size;
  const int*   inputs = (const int*)  d_in[0];
  const float* memory = (const float*)d_in[1];
  const float* W0 = (const float*)d_in[2];
  const float* U0 = (const float*)d_in[3];
  const float* b0 = (const float*)d_in[4];
  const float* W1 = (const float*)d_in[5];
  const float* U1 = (const float*)d_in[6];
  const float* b1 = (const float*)d_in[7];
  const float* Wm = (const float*)d_in[8];
  const float* Wa = (const float*)d_in[9];
  const float* Wf = (const float*)d_in[10];
  const float* bf = (const float*)d_in[11];
  float* out = (float*)d_out;

  char* ws = (char*)d_ws;
  size_t off = 0;
  auto alloc = [&](size_t bytes)->void*{
    void* p = ws + off; off += (bytes + 255) & ~(size_t)255; return p;
  };
  __bf16* memB = (__bf16*)alloc((size_t)B_*TIN_*DU_*2);  // 134 MB, L2-resident
  __bf16* W0vT = (__bf16*)alloc((size_t)2048*512*2);     // W0 rows V..V+DU, transposed
  __bf16* U0T  = (__bf16*)alloc((size_t)2048*512*2);
  __bf16* W1T  = (__bf16*)alloc((size_t)2048*512*2);
  __bf16* U1T  = (__bf16*)alloc((size_t)2048*512*2);
  __bf16* WmB  = (__bf16*)alloc((size_t)512*512*2);      // Wm itself == (Wm^T) transposed
  __bf16* WaT  = (__bf16*)alloc((size_t)512*1024*2);     // [512,1024]
  float*  c0   = (float*) alloc((size_t)B_*512*4);
  float*  c1   = (float*) alloc((size_t)B_*512*4);
  // read-buffers at t=0 (zeroed together):
  __bf16* h0Ba = (__bf16*)alloc((size_t)B_*512*2);
  __bf16* h1Ba = (__bf16*)alloc((size_t)B_*512*2);
  __bf16* atB  = (__bf16*)alloc((size_t)B_*512*2);
  // write-buffers (double-buffer partners):
  __bf16* h0Bb = (__bf16*)alloc((size_t)B_*512*2);
  __bf16* h1Bb = (__bf16*)alloc((size_t)B_*512*2);
  __bf16* qB   = (__bf16*)alloc((size_t)B_*512*2);
  __bf16* ctxB = (__bf16*)alloc((size_t)B_*512*2);

  // one-time prep (deterministic, re-run every call)
  k_conv<<<4096,256,0,stream>>>(memory, memB, (long)B_*TIN_*DU_);
  k_conv<<<1024,256,0,stream>>>(Wm, WmB, (long)512*512);
  k_conv_t<<<(512*2048+255)/256,256,0,stream>>>(W0, W0vT, 512, 2048, V_);
  k_conv_t<<<(512*2048+255)/256,256,0,stream>>>(U0, U0T, 512, 2048, 0);
  k_conv_t<<<(512*2048+255)/256,256,0,stream>>>(W1, W1T, 512, 2048, 0);
  k_conv_t<<<(512*2048+255)/256,256,0,stream>>>(U1, U1T, 512, 2048, 0);
  k_conv_t<<<(1024*512+255)/256,256,0,stream>>>(Wa, WaT, 1024, 512, 0);

  // zero initial carry: c0,c1 fp32 and h0Ba,h1Ba,atB bf16 (each contiguous)
  hipMemsetAsync(c0, 0, 2*(size_t)B_*512*4, stream);
  hipMemsetAsync(h0Ba, 0, 3*(size_t)B_*512*2, stream);

  dim3 blk(256);
  dim3 gL(8,4);   // lstm: 512 gate-d / 64, 256 rows / 64
  dim3 gS(2,4);   // plain: N=512, M=256
  for (int t = 0; t < TDEC_; ++t){
    __bf16* h0r = (t & 1) ? h0Bb : h0Ba;  __bf16* h0w = (t & 1) ? h0Ba : h0Bb;
    __bf16* h1r = (t & 1) ? h1Bb : h1Ba;  __bf16* h1w = (t & 1) ? h1Ba : h1Bb;
    // layer0: z0 = onehot@W0[0:V] (row gather) + attn@W0[V:] + h0@U0 + b0 ; gates fused
    k_lstm<<<gL,blk,0,stream>>>(atB, W0vT, h0r, U0T, b0,
                                W0, inputs + t, TDEC_, c0, h0w);
    // layer1: z1 = h0n@W1 + h1@U1 + b1 ; gates fused
    k_lstm<<<gL,blk,0,stream>>>(h0w, W1T, h1r, U1T, b1,
                                nullptr, nullptr, 0, c1, h1w);
    // q = h1n @ Wm^T   (keys_m never materialized)
    k_gemm<<<gS,blk,0,stream>>>(h1w, WmB, 512, nullptr, nullptr, 0, qB, 512);
    // scores/softmax/ctx fused, streaming bf16 memory from L2
    k_attn<<<256,512,0,stream>>>(qB, memB, ctxB);
    // attn_n = [h1n | ctx] @ Wa
    k_gemm<<<gS,blk,0,stream>>>(h1w, WaT, 1024, ctxB, WaT + 512, 1024, atB, 512);
    // logits -> out[:, t, :]
    k_logits<<<8,blk,0,stream>>>(atB, Wf, bf, out, t);
  }
}
